// GNNRegressor_5059471475393
// MI455X (gfx1250) — matrix operand order repaired and checked by reading, not compiled, not run on hardware
//
#include <hip/hip_runtime.h>

typedef __bf16 bf16_t;
typedef __attribute__((ext_vector_type(16))) __bf16 bf16x16;
typedef __attribute__((ext_vector_type(4)))  __bf16 bf16x4;
typedef __attribute__((ext_vector_type(8)))  float   f32x8;
typedef __attribute__((ext_vector_type(4)))  float   f32x4;
typedef __attribute__((ext_vector_type(4)))  unsigned int u32x4;

union Frag16 {
  bf16x16 v;
  u32x4          q[2];
  unsigned int   u[8];
  unsigned short s[16];
};

#define NN      32768
#define NE      262144
#define NG      512
#define HDIM    512
#define INDIM   2279
#define KPADIN  2304
#define NLAYER  6
#define TDIM    13

// ---------------------------------------------------------------------------
// Fragment loaders (CDNA5 wave32 WMMA; A/B k order is this code's contiguous one, not the ISA table's).
// A 16x32 bf16, lane = row M: K = 16*half..16*half+15 contiguous -> 2x ds_load_b128.
// B 32x16 bf16, lane = col N: K = 16*half..16*half+15 contiguous -> store B
//   column-major in LDS ([n][k]) so the frag is one 32B run -> 2x ds_load_b128.
// C/D f32 16x16: element r -> M = r + 8*half, N = lane&15.
// ---------------------------------------------------------------------------
__device__ inline bf16x16 load_afrag(const bf16_t* rowbase, int hl) {
  Frag16 f;
  f.q[0] = *(const u32x4*)(rowbase + 16 * hl);
  f.q[1] = *(const u32x4*)(rowbase + 16 * hl + 8);
  return f.v;
}

__device__ inline bf16x16 load_bfrag(const bf16_t* colbase, int hl) {
  Frag16 f;
  f.q[0] = *(const u32x4*)(colbase + 16 * hl);
  f.q[1] = *(const u32x4*)(colbase + 16 * hl + 8);
  return f.v;
}

// ---------------------------------------------------------------------------
// Main bf16 WMMA GEMM:  C[M,N] = A_f32[M,Kreal] * B_bf16[Kpad,N] (+bias)(+relu)
// Block 256 thr = 8 waves; tile 256(M) x 64(N); K-step 32.
// Each wave owns two 16-row strips -> every B fragment feeds 2 WMMAs.
// ALIGNED=true:  Kreal==Kpad, Kreal%4==0 -> branch-free vector staging.
// ALIGNED=false: scalar unguarded staging for full tiles + one guarded tail.
// M % 256 == 0 and N % 64 == 0 for all call sites.
// ---------------------------------------------------------------------------
#define GTM 256
#define GTN 64
#define GTK 32
#define ALD 40    // A LDS k-stride (elems): 80B rows, 16B-aligned
#define BLD 40    // B LDS k-stride per column (elems)

template <bool ALIGNED>
__global__ __launch_bounds__(256)
void gemm_bf16_wmma(const float* __restrict__ A, const bf16_t* __restrict__ B,
                    const float* __restrict__ bias, float* __restrict__ C,
                    int M, int Kreal, int Kpad, int N, int act) {
  __shared__ __align__(16) bf16_t As[GTM * ALD];   // 20 KB
  __shared__ __align__(16) bf16_t Bs[GTN * BLD];   // 5 KB, [n][k]
  const int tid  = threadIdx.x;
  const int wv   = tid >> 5;
  const int lane = tid & 31;
  const int hl   = lane >> 4;
  const int l16  = lane & 15;
  const int m0   = blockIdx.y * GTM;
  const int n0   = blockIdx.x * GTN;
  const int rr   = tid >> 3;          // 0..31 (stage row group)
  const int kc   = (tid & 7) * 4;     // 0,4,..,28 (stage k cols)

  f32x8 acc[2][4];
#pragma unroll
  for (int s = 0; s < 2; ++s)
#pragma unroll
    for (int t = 0; t < 4; ++t)
#pragma unroll
      for (int r = 0; r < 8; ++r) acc[s][t][r] = 0.0f;

  for (int k0 = 0; k0 < Kpad; k0 += GTK) {
    // ---- stage A tile 256x32 (fp32 -> bf16), 32 elems/thread ----
    if constexpr (ALIGNED) {
      // branch-free: float4 loads + packed cvt + 8B LDS stores
#pragma unroll
      for (int p = 0; p < 8; ++p) {
        const int row = p * 32 + rr;
        const f32x4 v4 = *(const f32x4*)(A + (size_t)(m0 + row) * Kreal + k0 + kc);
        *(bf16x4*)&As[row * ALD + kc] = __builtin_convertvector(v4, bf16x4);
      }
    } else {
      if (k0 + GTK <= Kreal) {
        // full tile: scalar loads (rows may be misaligned), no guards
#pragma unroll
        for (int p = 0; p < 8; ++p) {
          const int row = p * 32 + rr;
          const float* src = A + (size_t)(m0 + row) * Kreal + k0 + kc;
          f32x4 v4;
          v4.x = src[0]; v4.y = src[1]; v4.z = src[2]; v4.w = src[3];
          *(bf16x4*)&As[row * ALD + kc] = __builtin_convertvector(v4, bf16x4);
        }
      } else {
        // tail tile (at most once per block): guarded scalar loads
#pragma unroll
        for (int p = 0; p < 8; ++p) {
          const int row = p * 32 + rr;
          const float* src = A + (size_t)(m0 + row) * Kreal;
          bf16_t* dst = &As[row * ALD + kc];
#pragma unroll
          for (int j = 0; j < 4; ++j) {
            const int kg = k0 + kc + j;
            dst[j] = (bf16_t)((kg < Kreal) ? src[kg] : 0.0f);
          }
        }
      }
    }
    // ---- stage B tile 32x64 bf16, transposed into [n][k]; 8 elems/thread ----
    {
      const int kr = tid >> 3;          // 0..31
      const int nc = (tid & 7) * 8;     // 0,8,..,56
      const bf16_t* src = B + (size_t)(k0 + kr) * N + (n0 + nc);
#pragma unroll
      for (int j = 0; j < 8; ++j) Bs[(nc + j) * BLD + kr] = src[j];
    }
    if (k0 + GTK < Kpad)
      __builtin_prefetch(B + (size_t)(k0 + GTK) * N + n0 + tid, 0, 0);
    __syncthreads();

    const bf16x16 af0 = load_afrag(&As[(wv * 32 + l16) * ALD], hl);
    const bf16x16 af1 = load_afrag(&As[(wv * 32 + 16 + l16) * ALD], hl);
#pragma unroll
    for (int nt = 0; nt < 4; ++nt) {
      const bf16x16 bf = load_bfrag(&Bs[(nt * 16 + l16) * BLD], hl);
      acc[0][nt] = __builtin_amdgcn_wmma_f32_16x16x32_bf16(
          false, af0, false, bf, (short)0, acc[0][nt], false, false);
      acc[1][nt] = __builtin_amdgcn_wmma_f32_16x16x32_bf16(
          false, af1, false, bf, (short)0, acc[1][nt], false, false);
    }
    __syncthreads();
  }

#pragma unroll
  for (int s = 0; s < 2; ++s)
#pragma unroll
    for (int nt = 0; nt < 4; ++nt) {
      const int col = n0 + nt * 16 + l16;
      const float bv = bias ? bias[col] : 0.0f;
#pragma unroll
      for (int r = 0; r < 8; ++r) {
        const int row = m0 + wv * 32 + s * 16 + r + 8 * hl;
        float v = acc[s][nt][r] + bv;
        if (act) v = fmaxf(v, 0.0f);
        C[(size_t)row * N + col] = v;
      }
    }
}

// ---------------------------------------------------------------------------
// Batched attention: one block (4 waves) per (graph, head). 64 queries/keys,
// head dim 64. S = Q K^T / 8 -> softmax -> O = S V, all via bf16 WMMA.
// K staged natural [key][d]  (S-pass B-frag is contiguous along d).
// V staged transposed [d][key] (O-pass B-frag is contiguous along key).
// ---------------------------------------------------------------------------
__global__ __launch_bounds__(128)
void attention_wmma(const float* __restrict__ Q, const float* __restrict__ K,
                    const float* __restrict__ V, float* __restrict__ O) {
  __shared__ __align__(16) bf16_t Qs[64 * 72];   // [query][d]
  __shared__ __align__(16) bf16_t Ks[64 * 72];   // [key][d]
  __shared__ __align__(16) bf16_t VT[64 * 72];   // [d][key]
  __shared__ __align__(16) bf16_t Ss[64 * 72];   // [query][key], softmaxed
  const int g  = blockIdx.x >> 3;
  const int hh = blockIdx.x & 7;
  const int tid  = threadIdx.x;
  const int wv   = tid >> 5;
  const int lane = tid & 31;
  const int hl   = lane >> 4;
  const int l16  = lane & 15;

  // stage: 4 contiguous d per thread (float4 loads, packed cvt)
  for (int i = tid; i < 1024; i += 128) {
    const int row = i >> 4, d4 = (i & 15) * 4;
    const size_t gi = (size_t)(g * 64 + row) * HDIM + hh * 64 + d4;
    const f32x4 q4 = *(const f32x4*)(Q + gi);
    const f32x4 k4 = *(const f32x4*)(K + gi);
    const f32x4 v4 = *(const f32x4*)(V + gi);
    *(bf16x4*)&Qs[row * 72 + d4] = __builtin_convertvector(q4, bf16x4);
    *(bf16x4*)&Ks[row * 72 + d4] = __builtin_convertvector(k4, bf16x4);
    const bf16x4 vb = __builtin_convertvector(v4, bf16x4);
    VT[(d4 + 0) * 72 + row] = vb.x;
    VT[(d4 + 1) * 72 + row] = vb.y;
    VT[(d4 + 2) * 72 + row] = vb.z;
    VT[(d4 + 3) * 72 + row] = vb.w;
  }
  __syncthreads();

  const int m0 = wv * 16;
  f32x8 acc[4];
#pragma unroll
  for (int t = 0; t < 4; ++t)
#pragma unroll
    for (int r = 0; r < 8; ++r) acc[t][r] = 0.0f;

  // S = Q K^T  (B element (k=d, n=key) = Ks[key][d], contiguous in d)
#pragma unroll
  for (int ks = 0; ks < 64; ks += 32) {
    const bf16x16 af = load_afrag(&Qs[(m0 + l16) * 72 + ks], hl);
#pragma unroll
    for (int nt = 0; nt < 4; ++nt) {
      const bf16x16 bf = load_bfrag(&Ks[(nt * 16 + l16) * 72 + ks], hl);
      acc[nt] = __builtin_amdgcn_wmma_f32_16x16x32_bf16(
          false, af, false, bf, (short)0, acc[nt], false, false);
    }
  }

  // row softmax over 64 keys (4 tiles across lanes of the 16-lane group)
  const float scl = 0.125f;  // 1/sqrt(64)
#pragma unroll
  for (int r = 0; r < 8; ++r) {
    float v0 = acc[0][r] * scl, v1 = acc[1][r] * scl;
    float v2 = acc[2][r] * scl, v3 = acc[3][r] * scl;
    float mx = fmaxf(fmaxf(v0, v1), fmaxf(v2, v3));
#pragma unroll
    for (int m = 8; m >= 1; m >>= 1) mx = fmaxf(mx, __shfl_xor(mx, m, 32));
    const float e0 = __expf(v0 - mx), e1 = __expf(v1 - mx);
    const float e2 = __expf(v2 - mx), e3 = __expf(v3 - mx);
    float sm = e0 + e1 + e2 + e3;
#pragma unroll
    for (int m = 8; m >= 1; m >>= 1) sm += __shfl_xor(sm, m, 32);
    const float ri = 1.0f / sm;
    const int row = m0 + r + 8 * hl;
    Ss[row * 72 +  0 + l16] = (bf16_t)(e0 * ri);
    Ss[row * 72 + 16 + l16] = (bf16_t)(e1 * ri);
    Ss[row * 72 + 32 + l16] = (bf16_t)(e2 * ri);
    Ss[row * 72 + 48 + l16] = (bf16_t)(e3 * ri);
  }
  __syncthreads();

  // O = S V  (B element (k=key, n=d) = VT[d][key], contiguous in key)
#pragma unroll
  for (int t = 0; t < 4; ++t)
#pragma unroll
    for (int r = 0; r < 8; ++r) acc[t][r] = 0.0f;
#pragma unroll
  for (int ks = 0; ks < 64; ks += 32) {
    const bf16x16 af = load_afrag(&Ss[(m0 + l16) * 72 + ks], hl);
#pragma unroll
    for (int nt = 0; nt < 4; ++nt) {
      const bf16x16 bf = load_bfrag(&VT[(nt * 16 + l16) * 72 + ks], hl);
      acc[nt] = __builtin_amdgcn_wmma_f32_16x16x32_bf16(
          false, af, false, bf, (short)0, acc[nt], false, false);
    }
  }
#pragma unroll
  for (int nt = 0; nt < 4; ++nt) {
#pragma unroll
    for (int r = 0; r < 8; ++r) {
      const int node = g * 64 + m0 + r + 8 * hl;
      O[(size_t)node * HDIM + hh * 64 + nt * 16 + l16] = acc[nt][r];
    }
  }
}

// ---------------------------------------------------------------------------
// Prep / elementwise / graph kernels
// ---------------------------------------------------------------------------
__global__ void cvt_pad_kernel(const float* __restrict__ src, bf16_t* __restrict__ dst,
                               int Kreal, int Kpad, int N) {
  const long long idx = (long long)blockIdx.x * blockDim.x + threadIdx.x;
  if (idx >= (long long)Kpad * N) return;
  const int r = (int)(idx / N), c = (int)(idx % N);
  dst[idx] = (bf16_t)((r < Kreal) ? src[(long long)r * N + c] : 0.0f);
}

__global__ void bn_prep_kernel(const float* g, const float* b, const float* mean,
                               const float* var, float* sc, float* sh, int tot) {
  const int i = blockIdx.x * blockDim.x + threadIdx.x;
  if (i >= tot) return;
  const float s = g[i] * rsqrtf(var[i] + 1e-5f);
  sc[i] = s;
  sh[i] = b[i] - mean[i] * s;
}

__global__ void fill_kernel(float* p, float v, int n) {
  const int i = blockIdx.x * blockDim.x + threadIdx.x;
  if (i < n) p[i] = v;
}

__global__ void deg_edges_kernel(const int* __restrict__ ei, float* __restrict__ deg) {
  const int e = blockIdx.x * blockDim.x + threadIdx.x;
  if (e < NE) atomicAdd(&deg[ei[NE + e]], 1.0f);
}

__global__ void rsqrt_kernel(float* p, int n) {
  const int i = blockIdx.x * blockDim.x + threadIdx.x;
  if (i < n) p[i] = rsqrtf(p[i]);
}

// agg = bias + m * dis^2  (self-loop contribution, initializes agg); float4
__global__ void init_agg_kernel(const float* __restrict__ m, const float* __restrict__ dis,
                                const float* __restrict__ bias, float* __restrict__ agg) {
  const int i4 = (blockIdx.x * blockDim.x + threadIdx.x) * 4;  // < NN*HDIM
  const int v = i4 >> 9, f = i4 & 511;
  const float d2 = dis[v] * dis[v];
  const f32x4 mv = *(const f32x4*)(m + i4);
  const f32x4 bv = *(const f32x4*)(bias + f);
  f32x4 o;
  o.x = bv.x + mv.x * d2; o.y = bv.y + mv.y * d2;
  o.z = bv.z + mv.z * d2; o.w = bv.w + mv.w * d2;
  *(f32x4*)(agg + i4) = o;
}

// edge scatter: agg[dst] += m[src] * dis[src]*dis[dst]; 8 features/thread
__global__ void edge_scatter_kernel(const float* __restrict__ m, const int* __restrict__ ei,
                                    const float* __restrict__ dis, float* __restrict__ agg) {
  const int idx = blockIdx.x * blockDim.x + threadIdx.x;  // < NE*64
  const int e = idx >> 6;
  const int f0 = (idx & 63) << 3;
  const int s = ei[e], d = ei[NE + e];
  const float w = dis[s] * dis[d];
  const float4* mp = (const float4*)(m + (size_t)s * HDIM + f0);
  const float4 a = mp[0], b = mp[1];
  float* ap = agg + (size_t)d * HDIM + f0;
  atomicAdd(ap + 0, a.x * w); atomicAdd(ap + 1, a.y * w);
  atomicAdd(ap + 2, a.z * w); atomicAdd(ap + 3, a.w * w);
  atomicAdd(ap + 4, b.x * w); atomicAdd(ap + 5, b.y * w);
  atomicAdd(ap + 6, b.z * w); atomicAdd(ap + 7, b.w * w);
}

// h = relu(agg*sc + sh) + h   (fused BN + ReLU + residual); float4
__global__ void bn_res_kernel(const float* __restrict__ agg, const float* __restrict__ sc,
                              const float* __restrict__ sh, float* __restrict__ h) {
  const int i4 = (blockIdx.x * blockDim.x + threadIdx.x) * 4;  // < NN*HDIM
  const int f = i4 & 511;
  const f32x4 a = *(const f32x4*)(agg + i4);
  const f32x4 s = *(const f32x4*)(sc + f);
  const f32x4 t = *(const f32x4*)(sh + f);
  f32x4 hv = *(const f32x4*)(h + i4);
  hv.x += fmaxf(a.x * s.x + t.x, 0.0f);
  hv.y += fmaxf(a.y * s.y + t.y, 0.0f);
  hv.z += fmaxf(a.z * s.z + t.z, 0.0f);
  hv.w += fmaxf(a.w * s.w + t.w, 0.0f);
  *(f32x4*)(h + i4) = hv;
}

// pooled[g,f..f+3] = mean over 64 nodes; float4
__global__ void pool_kernel(const float* __restrict__ h, float* __restrict__ pooled) {
  const int i4 = (blockIdx.x * blockDim.x + threadIdx.x) * 4;  // < NG*HDIM
  const int g = i4 >> 9, f = i4 & 511;
  f32x4 s; s.x = 0.0f; s.y = 0.0f; s.z = 0.0f; s.w = 0.0f;
#pragma unroll 8
  for (int n = 0; n < 64; ++n) {
    const f32x4 v = *(const f32x4*)(h + (size_t)(g * 64 + n) * HDIM + f);
    s.x += v.x; s.y += v.y; s.z += v.z; s.w += v.w;
  }
  s.x *= (1.0f / 64.0f); s.y *= (1.0f / 64.0f);
  s.z *= (1.0f / 64.0f); s.w *= (1.0f / 64.0f);
  *(f32x4*)(pooled + i4) = s;
}

// out[g,t] = z[g,:] @ W2[:,t] + b2[t]   (N=13, tiny: plain dot)
__global__ void mlp2_kernel(const float* __restrict__ z, const float* __restrict__ W2,
                            const float* __restrict__ b2, float* __restrict__ out) {
  const int idx = blockIdx.x * blockDim.x + threadIdx.x;  // < NG*TDIM
  if (idx >= NG * TDIM) return;
  const int g = idx / TDIM, t = idx % TDIM;
  float s = b2[t];
  for (int k = 0; k < 256; ++k) s += z[g * 256 + k] * W2[k * TDIM + t];
  out[idx] = s;
}

// ---------------------------------------------------------------------------
extern "C" void kernel_launch(void* const* d_in, const int* in_sizes, int n_in,
                              void* d_out, int out_size, void* d_ws, size_t ws_size,
                              hipStream_t stream) {
  const float* x     = (const float*)d_in[0];
  const int*   ei    = (const int*)d_in[1];
  const float* W_in  = (const float*)d_in[3];
  const float* b_in  = (const float*)d_in[4];
  const float* W_gcn = (const float*)d_in[5];
  const float* b_gcn = (const float*)d_in[6];
  const float* bn_g  = (const float*)d_in[7];
  const float* bn_b  = (const float*)d_in[8];
  const float* bn_m  = (const float*)d_in[9];
  const float* bn_v  = (const float*)d_in[10];
  const float* W_q = (const float*)d_in[11]; const float* b_q = (const float*)d_in[12];
  const float* W_k = (const float*)d_in[13]; const float* b_k = (const float*)d_in[14];
  const float* W_v = (const float*)d_in[15]; const float* b_v = (const float*)d_in[16];
  const float* W_o = (const float*)d_in[17]; const float* b_o = (const float*)d_in[18];
  const float* W1  = (const float*)d_in[19]; const float* b1  = (const float*)d_in[20];
  const float* W2  = (const float*)d_in[21]; const float* b2  = (const float*)d_in[22];
  float* out = (float*)d_out;

  char* ws = (char*)d_ws;
  size_t off = 0;
  auto alloc = [&](size_t bytes) -> char* {
    char* p = ws + off;
    off += (bytes + 255) & ~(size_t)255;
    return p;
  };
  bf16_t* Win_b  = (bf16_t*)alloc((size_t)KPADIN * HDIM * 2);
  bf16_t* Wgcn_b = (bf16_t*)alloc((size_t)NLAYER * HDIM * HDIM * 2);
  bf16_t* Wq_b   = (bf16_t*)alloc((size_t)HDIM * HDIM * 2);
  bf16_t* Wk_b   = (bf16_t*)alloc((size_t)HDIM * HDIM * 2);
  bf16_t* Wv_b   = (bf16_t*)alloc((size_t)HDIM * HDIM * 2);
  bf16_t* Wo_b   = (bf16_t*)alloc((size_t)HDIM * HDIM * 2);
  bf16_t* W1_b   = (bf16_t*)alloc((size_t)HDIM * 256 * 2);
  float* sc      = (float*)alloc((size_t)NLAYER * HDIM * 4);
  float* sh      = (float*)alloc((size_t)NLAYER * HDIM * 4);
  float* dis     = (float*)alloc((size_t)NN * 4);
  float* h       = (float*)alloc((size_t)NN * HDIM * 4);
  float* mbuf    = (float*)alloc((size_t)NN * HDIM * 4);
  float* agg     = (float*)alloc((size_t)NN * HDIM * 4);
  float* vbuf    = (float*)alloc((size_t)NN * HDIM * 4);
  float* obuf    = (float*)alloc((size_t)NN * HDIM * 4);
  float* pooled  = (float*)alloc((size_t)NG * HDIM * 4);
  float* zbuf    = (float*)alloc((size_t)NG * 256 * 4);
  (void)ws_size; (void)in_sizes; (void)n_in; (void)out_size;

  const dim3 B256(256);
  auto cvt = [&](const float* s, bf16_t* d, int kr, int kp, int n) {
    const long long tot = (long long)kp * n;
    cvt_pad_kernel<<<(unsigned)((tot + 255) / 256), B256, 0, stream>>>(s, d, kr, kp, n);
  };
  auto gemm = [&](const float* A, const bf16_t* Bm, const float* bias, float* C,
                  int M, int Kreal, int Kpad, int N, int act) {
    const dim3 grid(N / GTN, M / GTM);
    if (Kreal == Kpad && (Kreal & 3) == 0)
      gemm_bf16_wmma<true><<<grid, B256, 0, stream>>>(A, Bm, bias, C, M, Kreal, Kpad, N, act);
    else
      gemm_bf16_wmma<false><<<grid, B256, 0, stream>>>(A, Bm, bias, C, M, Kreal, Kpad, N, act);
  };

  // weight conversion (bf16) + BN folding
  cvt(W_in,  Win_b,  INDIM, KPADIN, HDIM);
  cvt(W_gcn, Wgcn_b, NLAYER * HDIM, NLAYER * HDIM, HDIM);
  cvt(W_q, Wq_b, HDIM, HDIM, HDIM);
  cvt(W_k, Wk_b, HDIM, HDIM, HDIM);
  cvt(W_v, Wv_b, HDIM, HDIM, HDIM);
  cvt(W_o, Wo_b, HDIM, HDIM, HDIM);
  cvt(W1,  W1_b, HDIM, HDIM, 256);
  bn_prep_kernel<<<(NLAYER * HDIM) / 256, B256, 0, stream>>>(bn_g, bn_b, bn_m, bn_v, sc, sh, NLAYER * HDIM);

  // symmetric GCN norm: deg = indeg + 1 (self loop), dis = rsqrt(deg)
  fill_kernel<<<NN / 256, B256, 0, stream>>>(dis, 1.0f, NN);
  deg_edges_kernel<<<NE / 256, B256, 0, stream>>>(ei, dis);
  rsqrt_kernel<<<NN / 256, B256, 0, stream>>>(dis, NN);

  // input projection (big GEMM, K padded 2279->2304)
  gemm(x, Win_b, b_in, h, NN, INDIM, KPADIN, HDIM, 0);

  // GCN layers
  const int EW4 = NN * HDIM / 4 / 256;  // 16384 blocks, float4 elementwise
  const int SC  = NE * 64 / 256;        // 65536 blocks for edge scatter
  for (int i = 0; i < NLAYER; ++i) {
    gemm(h, Wgcn_b + (size_t)i * HDIM * HDIM, nullptr, mbuf, NN, HDIM, HDIM, HDIM, 0);
    init_agg_kernel<<<EW4, B256, 0, stream>>>(mbuf, dis, b_gcn + (size_t)i * HDIM, agg);
    edge_scatter_kernel<<<SC, B256, 0, stream>>>(mbuf, ei, dis, agg);
    bn_res_kernel<<<EW4, B256, 0, stream>>>(agg, sc + (size_t)i * HDIM, sh + (size_t)i * HDIM, h);
  }

  // attention: q/k/v projections then batched WMMA attention, then out-proj
  gemm(h, Wq_b, b_q, mbuf, NN, HDIM, HDIM, HDIM, 0);   // q
  gemm(h, Wk_b, b_k, agg,  NN, HDIM, HDIM, HDIM, 0);   // k
  gemm(h, Wv_b, b_v, vbuf, NN, HDIM, HDIM, HDIM, 0);   // v
  attention_wmma<<<NG * 8, dim3(128), 0, stream>>>(mbuf, agg, vbuf, obuf);
  gemm(obuf, Wo_b, b_o, h, NN, HDIM, HDIM, HDIM, 0);   // h = o @ W_o + b_o

  // pool + MLP head
  pool_kernel<<<(NG * HDIM / 4) / 256, B256, 0, stream>>>(h, pooled);
  gemm(pooled, W1_b, b1, zbuf, NG, HDIM, HDIM, 256, 1);
  mlp2_kernel<<<(NG * TDIM) / 256, B256, 0, stream>>>(zbuf, W2, b2, out);
}
